// LuenbergerLDS_58798102282592
// MI455X (gfx1250) — matrix operand order repaired and checked by reading, not compiled, not run on hardware
//
#include <hip/hip_runtime.h>
#include <hip/hip_bf16.h>
#include <stdint.h>

// Problem constants (reference: B=32, T=2048, N=512, M=512)
#define BB 32
#define TT 2048
#define NN 512
#define MM 512
#define K2 1024            // 2*N : [Re ; -Im] concatenated K dimension
#define ROWS (BB * TT)     // 65536 GEMM rows (t-major: row = t*B + b)

typedef __attribute__((ext_vector_type(16))) __bf16 bf16x16;
typedef __attribute__((ext_vector_type(8)))  float  v8f;

struct U32x8 { uint4 a, b; };   // 32 bytes == bf16x16

// ---------------- workspace layout (bytes) ----------------
// Wc   : float2[512*512]   @ 0          (2 MB, W -> becomes W^-1 in place)
// colk : float2[512]       @ 2097152    (pivot column snapshot)
// Abf  : __bf16[512*1024]  @ 2101248    (A^T, row m, cols k: [Ar ; Ai])
// S    : __bf16[65536*1024]@ 3149824    (134 MB, rows (t*B+b), cols [Re;-Im])
#define OFF_COLK 2097152
#define OFF_ABF  2101248
#define OFF_S    3149824

// ---------------------------------------------------------------------------
// CDNA5 async copy (memory -> LDS, ASYNCcnt-tracked), inline asm per
// CDNA5 ISA / 15.18.3 (portable across both toolchains).
// lds_off: wave-relative LDS byte offset (low 32 bits of flat shared pointer).
// ---------------------------------------------------------------------------
__device__ __forceinline__ void async_ld_b128(uint32_t lds_off, const void* g) {
  asm volatile("global_load_async_to_lds_b128 %0, %1, off"
               :: "v"(lds_off), "v"(g) : "memory");
}
#define S_WAIT_ASYNC(n) asm volatile("s_wait_asynccnt " #n ::: "memory")

// ---------------------------------------------------------------------------
// 1) Complex Gauss-Jordan inversion of W (in place, no pivoting)
// ---------------------------------------------------------------------------
__global__ void init_wc(const float* __restrict__ Wr, const float* __restrict__ Wi,
                        float2* __restrict__ Wc) {
  int i = blockIdx.x * blockDim.x + threadIdx.x;   // 262144 elems
  Wc[i] = make_float2(Wr[i], Wi[i]);
}

// one block of 512 threads: snapshot column k, scale row k by 1/p, a_kk = 1/p
__global__ void gj_prep(float2* __restrict__ Wc, float2* __restrict__ colk, int k) {
  int tid = threadIdx.x;                 // 0..511
  float2 p  = Wc[k * NN + k];
  float  d  = p.x * p.x + p.y * p.y;
  float2 pinv = make_float2(p.x / d, -p.y / d);
  float2 cv = Wc[tid * NN + k];          // column k snapshot (pre-update)
  float2 av = Wc[k * NN + tid];          // row k element
  __syncthreads();
  colk[tid] = cv;
  float2 r;
  if (tid == k) r = pinv;
  else { r.x = av.x * pinv.x - av.y * pinv.y;
         r.y = av.x * pinv.y + av.y * pinv.x; }
  Wc[k * NN + tid] = r;
}

// full-matrix rank-1 update for rows i != k
__global__ void gj_elim(float2* __restrict__ Wc, const float2* __restrict__ colk, int k) {
  int idx = blockIdx.x * blockDim.x + threadIdx.x; // 262144
  int i = idx >> 9, j = idx & (NN - 1);
  if (i == k) return;
  float2 ci = colk[i];
  if (j == k) {
    float2 pinv = Wc[k * NN + k];        // already 1/p
    float2 r;
    r.x = -(ci.x * pinv.x - ci.y * pinv.y);
    r.y = -(ci.x * pinv.y + ci.y * pinv.x);
    Wc[i * NN + k] = r;
  } else {
    float2 rk = Wc[k * NN + j];          // scaled row k (stable in this kernel)
    float2 a  = Wc[i * NN + j];
    a.x -= ci.x * rk.x - ci.y * rk.y;
    a.y -= ci.x * rk.y + ci.y * rk.x;
    Wc[i * NN + j] = a;
  }
}

// ---------------------------------------------------------------------------
// 2) A^T (bf16): Abf[m][j] = sum_i ReWinv[i][j]*C[i][m];  Abf[m][512+j] = Im part
// ---------------------------------------------------------------------------
__global__ void computeA_kernel(const float2* __restrict__ Winv,
                                const float* __restrict__ C,
                                __bf16* __restrict__ Abf) {
  int j = blockIdx.x * 256 + threadIdx.x;  // grid.x = 2 -> j in [0,512)
  int m = blockIdx.y;                      // 512
  float sr = 0.f, si = 0.f;
  for (int i = 0; i < NN; ++i) {
    float2 w = Winv[i * NN + j];           // coalesced over j
    float  c = C[i * MM + m];              // uniform -> scalar load
    sr += w.x * c;
    si += w.y * c;
  }
  Abf[(size_t)m * K2 + j]        = (__bf16)sr;
  Abf[(size_t)m * K2 + NN + j]   = (__bf16)si;
}

// ---------------------------------------------------------------------------
// 3) diagonal complex recurrence, f32 state, bf16 [Re ; -Im] output
// ---------------------------------------------------------------------------
__global__ void scan_kernel(const float* __restrict__ x,
                            const float* __restrict__ lre,
                            const float* __restrict__ lim,
                            __bf16* __restrict__ S) {
  int g = blockIdx.x * blockDim.x + threadIdx.x;   // 16384 threads
  int b = g >> 9;
  int n = g & (NN - 1);
  float mag = expf(lre[n]);
  float lr = mag * cosf(lim[n]);
  float li = mag * sinf(lim[n]);
  const float* xb = x + b * TT;
  float sr = 0.f, si = 0.f;
  for (int t = 0; t < TT; ++t) {
    float xv = xb[t];
    float nr = lr * sr - li * si + xv;
    float ni = lr * si + li * sr;
    sr = nr; si = ni;
    size_t row = (size_t)(t * BB + b) * K2;
    S[row + n]      = (__bf16)sr;
    S[row + NN + n] = (__bf16)(-si);       // fold the minus sign here
  }
}

// ---------------------------------------------------------------------------
// 4) GEMM: (65536 x 1024)bf16 @ (1024 x 512)bf16 -> f32, epilogue +x*D+Do,
//    output permuted to (b, t, m). Tile 128x128 per block, 8 waves of 32x64.
//    Double-buffered async (ASYNCcnt) fills: global -> LDS direct.
// ---------------------------------------------------------------------------
#define LDP 56   // LDS row stride (elems): 112B rows -> 16B aligned, 0-conflict b128

__global__ __launch_bounds__(256) void gemm_kernel(
    const __bf16* __restrict__ S, const __bf16* __restrict__ A,
    const float* __restrict__ x, const float* __restrict__ D,
    const float* __restrict__ Do, float* __restrict__ out) {
  __shared__ __bf16 lA[2][128 * LDP];
  __shared__ __bf16 lB[2][128 * LDP];
  const int tid  = threadIdx.x;
  const int lane = tid & 31;
  const int wid  = tid >> 5;         // wave32: 8 waves / 256 threads
  const int wm   = wid & 3;          // 4 row strips of 32
  const int wn   = wid >> 2;         // 2 col strips of 64
  const int rowBase = blockIdx.y * 128;
  const int colBase = blockIdx.x * 128;
  const int lrow = lane & 15;
  const int kh   = lane >> 4;

  // per-thread staging coordinates: two 8-elem chunks of the 128x32 tile
  const int idx0 = tid,       row0 = idx0 >> 2, c0 = (idx0 & 3) * 8;
  const int idx1 = tid + 256, row1 = idx1 >> 2, c1 = (idx1 & 3) * 8;
  const __bf16* gA0 = &S[(size_t)(rowBase + row0) * K2 + c0];
  const __bf16* gA1 = &S[(size_t)(rowBase + row1) * K2 + c1];
  const __bf16* gB0 = &A[(size_t)(colBase + row0) * K2 + c0];
  const __bf16* gB1 = &A[(size_t)(colBase + row1) * K2 + c1];
  // wave-relative LDS byte offsets (flat shared pointer: aperture is in the
  // high 32 bits, low 32 bits are the LDS offset)
  const uint32_t e0 = (uint32_t)(row0 * LDP + c0) * 2u;
  const uint32_t e1 = (uint32_t)(row1 * LDP + c1) * 2u;
  uint32_t baseA[2] = { (uint32_t)(uintptr_t)&lA[0][0], (uint32_t)(uintptr_t)&lA[1][0] };
  uint32_t baseB[2] = { (uint32_t)(uintptr_t)&lB[0][0], (uint32_t)(uintptr_t)&lB[1][0] };

  auto issue_stage = [&](int buf, int k0) {
    async_ld_b128(baseA[buf] + e0, gA0 + k0);   // 4 ASYNCcnt ops per wave
    async_ld_b128(baseA[buf] + e1, gA1 + k0);
    async_ld_b128(baseB[buf] + e0, gB0 + k0);
    async_ld_b128(baseB[buf] + e1, gB1 + k0);
  };

  v8f acc[2][4] = {};

  issue_stage(0, 0);                    // prologue fill
  for (int kt = 0; kt < K2 / 32; ++kt) {
    const int cur = kt & 1;
    if (kt + 1 < K2 / 32) {
      issue_stage(cur ^ 1, (kt + 1) * 32);   // overlap next fill with compute
      S_WAIT_ASYNC(4);                       // my stage-kt fills done (in order)
    } else {
      S_WAIT_ASYNC(0);
    }
    __syncthreads();                         // all waves' fills visible

    // ---- A fragments (16-bit A 16x32 layout: halves interleave K 0-7|8-15, +16) ----
    bf16x16 af[2];
#pragma unroll
    for (int mi = 0; mi < 2; ++mi) {
      int row = wm * 32 + mi * 16 + lrow;
      U32x8 tmp;
      tmp.a = *(const uint4*)&lA[cur][row * LDP + kh * 8];
      tmp.b = *(const uint4*)&lA[cur][row * LDP + 16 + kh * 8];
      af[mi] = __builtin_bit_cast(bf16x16, tmp);
    }
    // ---- B fragments (16-bit B: same K order as A: halves interleave K 0-7|8-15, +16) ----
    bf16x16 bfr[4];
#pragma unroll
    for (int ni = 0; ni < 4; ++ni) {
      int col = wn * 64 + ni * 16 + lrow;
      U32x8 tmp;
      tmp.a = *(const uint4*)&lB[cur][col * LDP + kh * 8];
      tmp.b = *(const uint4*)&lB[cur][col * LDP + 16 + kh * 8];
      bfr[ni] = __builtin_bit_cast(bf16x16, tmp);
    }
#pragma unroll
    for (int mi = 0; mi < 2; ++mi)
#pragma unroll
      for (int ni = 0; ni < 4; ++ni)
        acc[mi][ni] = __builtin_amdgcn_wmma_f32_16x16x32_bf16(
            false, af[mi], false, bfr[ni], (short)0, acc[mi][ni], false, false);
    __syncthreads();      // buffer `cur` is re-filled at iteration kt+1
  }

  // ---- epilogue: + x*D + Do, permute (t*B+b) rows -> out[b][t][m] ----
#pragma unroll
  for (int mi = 0; mi < 2; ++mi) {
    float xr[8];
#pragma unroll
    for (int v = 0; v < 8; ++v) {
      int r = rowBase + wm * 32 + mi * 16 + v + (kh << 3);
      int b = r & (BB - 1);
      int t = r >> 5;
      xr[v] = x[b * TT + t];
    }
#pragma unroll
    for (int ni = 0; ni < 4; ++ni) {
      int col = colBase + wn * 64 + ni * 16 + lrow;
      float Dv = D[col], Dov = Do[col];
#pragma unroll
      for (int v = 0; v < 8; ++v) {
        int r = rowBase + wm * 32 + mi * 16 + v + (kh << 3);
        int b = r & (BB - 1);
        int t = r >> 5;
        out[((size_t)b * TT + t) * MM + col] = acc[mi][ni][v] + xr[v] * Dv + Dov;
      }
    }
  }
}

// ---------------------------------------------------------------------------
extern "C" void kernel_launch(void* const* d_in, const int* in_sizes, int n_in,
                              void* d_out, int out_size, void* d_ws, size_t ws_size,
                              hipStream_t stream) {
  (void)in_sizes; (void)n_in; (void)out_size; (void)ws_size;
  const float* x   = (const float*)d_in[0];
  const float* lre = (const float*)d_in[1];
  const float* lim = (const float*)d_in[2];
  const float* Wr  = (const float*)d_in[3];
  const float* Wi  = (const float*)d_in[4];
  const float* C   = (const float*)d_in[5];
  const float* D   = (const float*)d_in[6];
  const float* Do  = (const float*)d_in[7];
  float* out = (float*)d_out;

  char* ws = (char*)d_ws;
  float2* Wc   = (float2*)(ws);
  float2* colk = (float2*)(ws + OFF_COLK);
  __bf16* Abf  = (__bf16*)(ws + OFF_ABF);
  __bf16* S    = (__bf16*)(ws + OFF_S);

  // 1) W -> W^-1 (complex, in place)
  init_wc<<<1024, 256, 0, stream>>>(Wr, Wi, Wc);
  for (int k = 0; k < NN; ++k) {
    gj_prep<<<1, 512, 0, stream>>>(Wc, colk, k);
    gj_elim<<<1024, 256, 0, stream>>>(Wc, colk, k);
  }
  // 2) fused weights A^T = [Re(W^-1)^T C ; Im(W^-1)^T C] in bf16
  computeA_kernel<<<dim3(2, 512), 256, 0, stream>>>(Wc, C, Abf);
  // 3) recurrence -> S (bf16 [Re ; -Im])
  scan_kernel<<<64, 256, 0, stream>>>(x, lre, lim, S);
  // 4) big WMMA GEMM with async double-buffered fills + fused epilogue
  gemm_kernel<<<dim3(MM / 128, ROWS / 128), 256, 0, stream>>>(S, Abf, x, D, Do, out);
}
